// FLOUNDERLRecurrentAgentLevel1_35364760715760
// MI455X (gfx1250) — matrix operand order repaired and checked by reading, not compiled, not run on hardware
//
#include <hip/hip_runtime.h>
#include <stdint.h>

#define WAVES 2
#define NBATCH 4096
#define TLEN 64
#define DIN 512
#define HDIM 64
#define NP 45
#define NPAD 48

typedef __attribute__((ext_vector_type(16))) __bf16 v16bf;
typedef __attribute__((ext_vector_type(8)))  float  v8f;

union ABf { v16bf v; uint32_t u[8]; uint4 q[2]; };

// round-half-up bf16 pack: 2x v_add_u32 + 1x v_perm_b32
__device__ __forceinline__ uint32_t pk_bf16(float a, float b) {
  uint32_t ua = __float_as_uint(a) + 0x8000u;
  uint32_t ub = __float_as_uint(b) + 0x8000u;
  return __builtin_amdgcn_perm(ua, ub, 0x03020706u);
}
__device__ __forceinline__ unsigned short bf16_1(float a) {
  return (unsigned short)((__float_as_uint(a) + 0x8000u) >> 16);
}
__device__ __forceinline__ float sigm(float x) { return 1.f / (1.f + __expf(-x)); }
__device__ __forceinline__ float tanh_f(float x) { float e = __expf(2.f * x); return 1.f - 2.f / (e + 1.f); }
__device__ __forceinline__ v8f splat8(float b) {
  v8f r;
  #pragma unroll
  for (int i = 0; i < 8; ++i) r[i] = b;
  return r;
}
#define WAIT_DS() asm volatile("s_wait_dscnt 0" ::: "memory")

__global__ __launch_bounds__(WAVES * 32)
__attribute__((amdgpu_num_vgpr(256)))
void gru_fused_kernel(const float* __restrict__ x, const float* __restrict__ h0p,
                      const float* __restrict__ epsp, const float* __restrict__ Wfc1,
                      const float* __restrict__ bfc1, const float* __restrict__ Wih,
                      const float* __restrict__ Whh, const float* __restrict__ bih,
                      const float* __restrict__ bhh, const float* __restrict__ Wout,
                      const float* __restrict__ bout, float* __restrict__ out) {
  // bf16 row-major weight staging in LDS (CDNA5: 320KB/WGP)
  __shared__ unsigned short sWfc1[HDIM * DIN];      // 64 KB
  __shared__ unsigned short sWih[3 * HDIM * HDIM];  // 24 KB
  __shared__ unsigned short sWhh[3 * HDIM * HDIM];  // 24 KB
  __shared__ unsigned short sWout[NPAD * HDIM];     // 6 KB
  __shared__ unsigned short sEnc[WAVES][16 * HDIM]; // per-wave A staging
  __shared__ unsigned short sH[WAVES][16 * HDIM];

  const int tid = threadIdx.x;
  for (int i = tid; i < HDIM * DIN; i += WAVES * 32) sWfc1[i] = bf16_1(Wfc1[i]);
  for (int i = tid; i < 3 * HDIM * HDIM; i += WAVES * 32) {
    sWih[i] = bf16_1(Wih[i]);
    sWhh[i] = bf16_1(Whh[i]);
  }
  for (int i = tid; i < NPAD * HDIM; i += WAVES * 32)
    sWout[i] = (i < NP * HDIM) ? bf16_1(Wout[i]) : (unsigned short)0;
  __syncthreads();

  const int lane = tid & 31;
  const int wv   = tid >> 5;
  const int ln   = lane & 15;
  const int hi   = lane >> 4;
  const int b0   = (blockIdx.x * WAVES + wv) * 16;
  unsigned short* estage = sEnc[wv];
  unsigned short* hstage = sH[wv];

  // per-lane bias / epsilon preloads (column-indexed: constant over t)
  float bias_enc[4], bias_rz[8], bias_in[4], bias_hn[4], bias_out3[3], epsv[8];
  #pragma unroll
  for (int nt = 0; nt < 4; ++nt) {
    bias_enc[nt] = bfc1[nt * 16 + ln];
    bias_in[nt]  = bih[128 + nt * 16 + ln];
    bias_hn[nt]  = bhh[128 + nt * 16 + ln];
  }
  #pragma unroll
  for (int q = 0; q < 8; ++q) { int c = q * 16 + ln; bias_rz[q] = bih[c] + bhh[c]; }
  #pragma unroll
  for (int nt = 0; nt < 3; ++nt) { int c = nt * 16 + ln; bias_out3[nt] = (c < NP) ? bout[c] : 0.f; }
  #pragma unroll
  for (int v = 0; v < 8; ++v) epsv[v] = epsp[b0 + v + 8 * hi];

  // h0 in C-layout registers + bf16 staging for first gh GEMM
  v8f hold[4];
  #pragma unroll
  for (int nt = 0; nt < 4; ++nt) {
    #pragma unroll
    for (int v = 0; v < 8; ++v) {
      float h = h0p[(size_t)(b0 + v + 8 * hi) * HDIM + nt * 16 + ln];
      hold[nt][v] = h;
      hstage[(v + 8 * hi) * HDIM + nt * 16 + ln] = bf16_1(h);
    }
  }
  WAIT_DS();

  const float* xrow0 = x + (size_t)(b0 + ln) * TLEN * DIN;
  float* xout = out;
  float* hout = out + (size_t)NBATCH * TLEN * NP;

  #pragma unroll 1
  for (int t = 0; t < TLEN; ++t) {
    const float* xr = xrow0 + (size_t)t * DIN;
    if (t + 1 < TLEN) {
      __builtin_prefetch(xrow0 + (size_t)(t + 1) * DIN + hi * 256, 0, 3);
      __builtin_prefetch(xrow0 + (size_t)(t + 1) * DIN + hi * 256 + 128, 0, 3);
    }

    // ---- enc = relu(x @ Wfc1^T + b): 16x64, K=512 ----
    v8f eacc[4];
    #pragma unroll
    for (int nt = 0; nt < 4; ++nt) eacc[nt] = splat8(bias_enc[nt]);
    #pragma unroll
    for (int kt = 0; kt < 16; ++kt) {
      const float4* p0 = (const float4*)(xr + kt * 32 + hi * 8);
      float4 f0 = p0[0], f1 = p0[1], f2 = p0[4], f3 = p0[5];
      // batch all 4 B fragments first: one dscnt wait feeds 4 back-to-back WMMAs
      ABf b4[4];
      #pragma unroll
      for (int nt = 0; nt < 4; ++nt) {
        const uint4* bp = (const uint4*)((const char*)sWfc1 +
                          ((nt * 16 + ln) * DIN + kt * 32 + hi * 8) * 2);
        b4[nt].q[0] = bp[0]; b4[nt].q[1] = bp[2];
      }
      ABf a;
      a.u[0] = pk_bf16(f0.x, f0.y); a.u[1] = pk_bf16(f0.z, f0.w);
      a.u[2] = pk_bf16(f1.x, f1.y); a.u[3] = pk_bf16(f1.z, f1.w);
      a.u[4] = pk_bf16(f2.x, f2.y); a.u[5] = pk_bf16(f2.z, f2.w);
      a.u[6] = pk_bf16(f3.x, f3.y); a.u[7] = pk_bf16(f3.z, f3.w);
      #pragma unroll
      for (int nt = 0; nt < 4; ++nt) {
        eacc[nt] = __builtin_amdgcn_wmma_f32_16x16x32_bf16(
            false, a.v, false, b4[nt].v, (short)0, eacc[nt], false, false);
      }
    }
    #pragma unroll
    for (int nt = 0; nt < 4; ++nt) {
      #pragma unroll
      for (int v = 0; v < 8; ++v) {
        float e = eacc[nt][v];
        e = e > 0.f ? e : 0.f;
        estage[(v + 8 * hi) * HDIM + nt * 16 + ln] = bf16_1(e);
      }
    }
    WAIT_DS();

    // ---- gi = enc @ Wih^T ; gh = h @ Whh^T : 16x192 each, K=64 ----
    v8f rz[8], inn[4], hn[4];
    #pragma unroll
    for (int q = 0; q < 8; ++q) rz[q] = splat8(bias_rz[q]);
    #pragma unroll
    for (int nt = 0; nt < 4; ++nt) { inn[nt] = splat8(bias_in[nt]); hn[nt] = splat8(bias_hn[nt]); }
    #pragma unroll
    for (int kt = 0; kt < 2; ++kt) {
      ABf a;
      const uint4* ap = (const uint4*)((const char*)estage + (ln * HDIM + kt * 32 + hi * 8) * 2);
      a.q[0] = ap[0]; a.q[1] = ap[2];
      #pragma unroll
      for (int grp = 0; grp < 3; ++grp) {
        ABf b4[4];
        #pragma unroll
        for (int i = 0; i < 4; ++i) {
          int g = grp * 4 + i;
          const uint4* bp = (const uint4*)((const char*)sWih +
                            ((g * 16 + ln) * HDIM + kt * 32 + hi * 8) * 2);
          b4[i].q[0] = bp[0]; b4[i].q[1] = bp[2];
        }
        #pragma unroll
        for (int i = 0; i < 4; ++i) {
          int g = grp * 4 + i;
          v8f& acc = (g < 8) ? rz[g] : inn[g - 8];
          acc = __builtin_amdgcn_wmma_f32_16x16x32_bf16(
              false, a.v, false, b4[i].v, (short)0, acc, false, false);
        }
      }
    }
    #pragma unroll
    for (int kt = 0; kt < 2; ++kt) {
      ABf a;
      const uint4* ap = (const uint4*)((const char*)hstage + (ln * HDIM + kt * 32 + hi * 8) * 2);
      a.q[0] = ap[0]; a.q[1] = ap[2];
      #pragma unroll
      for (int grp = 0; grp < 3; ++grp) {
        ABf b4[4];
        #pragma unroll
        for (int i = 0; i < 4; ++i) {
          int g = grp * 4 + i;
          const uint4* bp = (const uint4*)((const char*)sWhh +
                            ((g * 16 + ln) * HDIM + kt * 32 + hi * 8) * 2);
          b4[i].q[0] = bp[0]; b4[i].q[1] = bp[2];
        }
        #pragma unroll
        for (int i = 0; i < 4; ++i) {
          int g = grp * 4 + i;
          v8f& acc = (g < 8) ? rz[g] : hn[g - 8];
          acc = __builtin_amdgcn_wmma_f32_16x16x32_bf16(
              false, a.v, false, b4[i].v, (short)0, acc, false, false);
        }
      }
    }

    // ---- GRU gates: fully register-aligned in C-layout ----
    #pragma unroll
    for (int nt = 0; nt < 4; ++nt) {
      #pragma unroll
      for (int v = 0; v < 8; ++v) {
        float r = sigm(rz[nt][v]);
        float z = sigm(rz[4 + nt][v]);
        float n = tanh_f(inn[nt][v] + r * hn[nt][v]);
        float h = (1.f - z) * n + z * hold[nt][v];
        hold[nt][v] = h;
        hstage[(v + 8 * hi) * HDIM + nt * 16 + ln] = bf16_1(h);
        hout[((size_t)(b0 + v + 8 * hi) * TLEN + t) * HDIM + nt * 16 + ln] = h;
      }
    }
    WAIT_DS();

    // ---- logits = h @ Wout^T : 16x48 (45 valid), K=64 ----
    v8f la[3];
    #pragma unroll
    for (int nt = 0; nt < 3; ++nt) la[nt] = splat8(bias_out3[nt]);
    #pragma unroll
    for (int kt = 0; kt < 2; ++kt) {
      ABf a;
      const uint4* ap = (const uint4*)((const char*)hstage + (ln * HDIM + kt * 32 + hi * 8) * 2);
      a.q[0] = ap[0]; a.q[1] = ap[2];
      ABf b3[3];
      #pragma unroll
      for (int nt = 0; nt < 3; ++nt) {
        const uint4* bp = (const uint4*)((const char*)sWout +
                          ((nt * 16 + ln) * HDIM + kt * 32 + hi * 8) * 2);
        b3[nt].q[0] = bp[0]; b3[nt].q[1] = bp[2];
      }
      #pragma unroll
      for (int nt = 0; nt < 3; ++nt) {
        la[nt] = __builtin_amdgcn_wmma_f32_16x16x32_bf16(
            false, a.v, false, b3[nt].v, (short)0, la[nt], false, false);
      }
    }

    // ---- softmax + epsilon mix; row-sum via wave32 xor-shuffles ----
    #pragma unroll
    for (int v = 0; v < 8; ++v) {
      float s = 0.f;
      #pragma unroll
      for (int nt = 0; nt < 3; ++nt) {
        int c = nt * 16 + ln;
        float pv = (c < NP) ? __expf(la[nt][v]) : 0.f;
        la[nt][v] = pv;
        s += pv;
      }
      s += __shfl_xor(s, 1, 32);
      s += __shfl_xor(s, 2, 32);
      s += __shfl_xor(s, 4, 32);
      s += __shfl_xor(s, 8, 32);
      if (s <= 4.8789097e-18f) s = 1.f;
      float e = epsv[v];
      float scale = (1.f - e) / s;
      float addc = e * (1.f / 45.f);
      size_t ob = ((size_t)(b0 + v + 8 * hi) * TLEN + t) * NP;
      #pragma unroll
      for (int nt = 0; nt < 3; ++nt) {
        int c = nt * 16 + ln;
        if (c < NP) xout[ob + c] = la[nt][v] * scale + addc;
      }
    }
  }
}

extern "C" void kernel_launch(void* const* d_in, const int* in_sizes, int n_in,
                              void* d_out, int out_size, void* d_ws, size_t ws_size,
                              hipStream_t stream) {
  const float* x    = (const float*)d_in[0];
  const float* h0   = (const float*)d_in[1];
  const float* eps  = (const float*)d_in[2];
  const float* Wfc1 = (const float*)d_in[3];
  const float* bfc1 = (const float*)d_in[4];
  const float* Wih  = (const float*)d_in[5];
  const float* Whh  = (const float*)d_in[6];
  const float* bih  = (const float*)d_in[7];
  const float* bhh  = (const float*)d_in[8];
  const float* Wout = (const float*)d_in[9];
  const float* bout = (const float*)d_in[10];
  float* out = (float*)d_out;

  dim3 grid(NBATCH / (16 * WAVES));
  dim3 block(WAVES * 32);
  hipLaunchKernelGGL(gru_fused_kernel, grid, block, 0, stream,
                     x, h0, eps, Wfc1, bfc1, Wih, Whh, bih, bhh, Wout, bout, out);
}
